// AVWGCN_21715354649729
// MI455X (gfx1250) — matrix operand order repaired and checked by reading, not compiled, not run on hardware
//
#include <hip/hip_runtime.h>
#include <hip/hip_bf16.h>

// ---------------------------------------------------------------------------
// AVWGCN on MI455X (gfx1250), wave32 + v_wmma_f32_16x16x32_bf16.
//
// Math restructuring:
//   T2 = 2A·A - I is never materialized:
//     Y1 = A @ Xr            (Xr = X permuted to [N, B*C] = [4096,1024])
//     Y2 = A @ Y1
//     X_G[k=0]=X, X_G[k=1]=Y1, X_G[k=2]=2*Y2 - X
//   -> two 4096x4096x1024 bf16 GEMMs (~68 GFLOP) instead of 137 GFLOP A@A.
//   W = E @ weights_pool is ALSO a WMMA GEMM ([4096x16]@[16x12288], K padded
//   to 32), avoiding 3.2 GB of per-node L2 pool re-reads (96 MB HBM instead).
// A (bf16, 32MB) and pool are L2-resident (192MB L2) -> WMMA-bound.
// Main GEMM wave tile = 32x128 (2 A-frags share each B-frag): 16 WMMAs per
// 20 b128 loads per k-step, keeping the WMMA pipe (the scarce resource) fed.
// ---------------------------------------------------------------------------

typedef __attribute__((ext_vector_type(16))) __bf16          v16bf;
typedef __attribute__((ext_vector_type(8)))  float           v8f;
typedef __attribute__((ext_vector_type(8)))  unsigned short  v8u16;
typedef __attribute__((ext_vector_type(16))) unsigned short  v16u16;

union BF16Frag {
    v16bf  bf;
    v16u16 u16;
    v8u16  h[2];
};

__device__ __forceinline__ unsigned short f2bf(float f) {
    union { float f; unsigned int u; } v; v.f = f;
    unsigned int u = v.u;
    unsigned int r = (u + 0x7FFFu + ((u >> 16) & 1u)) >> 16;  // RNE truncate
    return (unsigned short)r;
}

#define N_NODES 4096
#define EMB_D   16
#define BATCH   16
#define CH      64
#define KORD    3
#define RDIM    (KORD * CH)    // 192
#define WCOLS   (CH * RDIM)    // 12288  (j' = o*192 + r)
#define JCOLS   (BATCH * CH)   // 1024

// ---------------------------------------------------------------------------
// Kernel 1: A = softmax(relu(E E^T)) per row, stored bf16 [4096][4096].
// ---------------------------------------------------------------------------
__global__ __launch_bounds__(256) void row_softmax_bf16(
        const float* __restrict__ E, unsigned short* __restrict__ Abf) {
    const int n   = blockIdx.x;
    const int tid = threadIdx.x;
    __shared__ float sRow[N_NODES];
    __shared__ float sRed[256];
    __shared__ float sEn[EMB_D];
    if (tid < EMB_D) sEn[tid] = E[n * EMB_D + tid];
    __syncthreads();
    float en[EMB_D];
    #pragma unroll
    for (int d = 0; d < EMB_D; ++d) en[d] = sEn[d];

    float lmax = 0.0f;                           // relu() >= 0
    for (int m = tid; m < N_NODES; m += 256) {
        const float* em = E + m * EMB_D;
        float s = 0.f;
        #pragma unroll
        for (int d = 0; d < EMB_D; ++d) s += en[d] * em[d];
        s = s > 0.f ? s : 0.f;
        sRow[m] = s;
        lmax = fmaxf(lmax, s);
    }
    sRed[tid] = lmax; __syncthreads();
    for (int off = 128; off > 0; off >>= 1) {
        if (tid < off) sRed[tid] = fmaxf(sRed[tid], sRed[tid + off]);
        __syncthreads();
    }
    const float rmax = sRed[0];
    __syncthreads();

    float lsum = 0.f;
    for (int m = tid; m < N_NODES; m += 256) {
        float v = __expf(sRow[m] - rmax);
        sRow[m] = v;
        lsum += v;
    }
    sRed[tid] = lsum; __syncthreads();
    for (int off = 128; off > 0; off >>= 1) {
        if (tid < off) sRed[tid] += sRed[tid + off];
        __syncthreads();
    }
    const float inv = 1.0f / sRed[0];
    for (int m = tid; m < N_NODES; m += 256)
        Abf[(size_t)n * N_NODES + m] = f2bf(sRow[m] * inv);
}

// ---------------------------------------------------------------------------
// Kernel 2: XrT[j][m] = bf16(X[b][m][c]), j = b*64+c. Transposed so the GEMM
// B-fragments are contiguous 32B per-lane loads.
// ---------------------------------------------------------------------------
__global__ __launch_bounds__(256) void permute_x(
        const float* __restrict__ X, unsigned short* __restrict__ XrT) {
    const int idx = blockIdx.x * 256 + threadIdx.x;   // j*4096 + m
    const int j = idx >> 12;
    const int m = idx & 4095;
    const int b = j >> 6;
    const int c = j & 63;
    XrT[idx] = f2bf(X[((size_t)b * N_NODES + m) * CH + c]);
}

// ---------------------------------------------------------------------------
// Kernel 3: poolT[j'][d] = bf16(pool[d][r][o]), j' = o*192 + r.
// ---------------------------------------------------------------------------
__global__ __launch_bounds__(256) void transpose_pool(
        const float* __restrict__ pool, unsigned short* __restrict__ poolT) {
    const int idx = blockIdx.x * 256 + threadIdx.x;   // j'*16 + d
    const int d  = idx & 15;
    const int jp = idx >> 4;
    const int o  = jp / RDIM;
    const int r  = jp - o * RDIM;
    poolT[idx] = f2bf(pool[(size_t)d * WCOLS + r * CH + o]);
}

// ---------------------------------------------------------------------------
// Kernel 4: WbfT[n][o*192+r] = bf16( sum_d E[n,d] * pool[d,r,o] ) as a WMMA
// GEMM [4096x16] @ [16x12288], K zero-padded 16 -> 32.
// Block 256 thr = 8 waves; block tile 16 rows x 1024 cols; wave = 8 col tiles.
// ---------------------------------------------------------------------------
__global__ __launch_bounds__(256) void wgemm_bf16(
        const float* __restrict__ E, const unsigned short* __restrict__ poolT,
        unsigned short* __restrict__ WbfT) {
    const int lane = threadIdx.x & 31;
    const int wave = threadIdx.x >> 5;
    const int half = lane >> 4;
    const int l15  = lane & 15;
    const int rowBase = blockIdx.y * 16;
    const int colBase = blockIdx.x * 1024 + wave * 128;

    const v16u16 vz = {};

    // A fragment: E rows, K = d0..15 real, K16..31 zero-padded.
    BF16Frag a;
    {
        const float* ep = E + (rowBase + l15) * EMB_D + half * 8;
        v8u16 lo;
        #pragma unroll
        for (int j = 0; j < 8; ++j) lo[j] = f2bf(ep[j]);
        a.h[0] = lo;                       // K = half*8 .. +7
        a.u16.hi = vz.lo;                  // zero upper (K 16..31)
        a.h[1] = (v8u16)(vz.lo);
    }

    v8f acc[8] = {};
    #pragma unroll
    for (int ct = 0; ct < 8; ++ct) {
        const int col = colBase + ct * 16 + l15;
        BF16Frag b;
        b.u16 = vz;
        // K = half*8 .. +7 (real d), K 16..31 zero
        b.h[0] = *(const v8u16*)(poolT + (size_t)col * EMB_D + half * 8);
        acc[ct] = __builtin_amdgcn_wmma_f32_16x16x32_bf16(
            false, a.bf, false, b.bf, (short)0, acc[ct], false, false);
    }

    #pragma unroll
    for (int ct = 0; ct < 8; ++ct) {
        const int col = colBase + ct * 16 + l15;
        #pragma unroll
        for (int r = 0; r < 8; ++r) {
            const int row = rowBase + r + half * 8;
            WbfT[(size_t)row * WCOLS + col] = f2bf(acc[ct][r]);
        }
    }
}

// ---------------------------------------------------------------------------
// Kernel 5: Y[4096][1024] = A_bf16 @ Rhs  (Rhs given transposed, bf16
// [1024][4096]). Also writes YbfT (bf16 transposed) to feed the next pass.
// Block 128 thr = 4 waves; tile 128x128; wave = 32 rows x 8 col-tiles.
// ---------------------------------------------------------------------------
__global__ __launch_bounds__(128) void gemm_bf16(
        const unsigned short* __restrict__ Abf,
        const unsigned short* __restrict__ RhsT,
        float* __restrict__ Y,
        unsigned short* __restrict__ YbfT) {
    const int lane = threadIdx.x & 31;
    const int wave = threadIdx.x >> 5;
    const int half = lane >> 4;
    const int l15  = lane & 15;
    const int rowBase = blockIdx.y * 128 + wave * 32;
    const int colBase = blockIdx.x * 128;

    v8f acc0[8] = {};
    v8f acc1[8] = {};

    const unsigned short* aRow0 = Abf + (size_t)(rowBase + l15) * N_NODES + half * 16;
    const unsigned short* aRow1 = aRow0 + (size_t)16 * N_NODES;
    for (int k0 = 0; k0 < N_NODES; k0 += 32) {
        BF16Frag a0, a1;
        a0.h[0] = *(const v8u16*)(aRow0 + k0);
        a0.h[1] = *(const v8u16*)(aRow0 + k0 + 8);
        a1.h[0] = *(const v8u16*)(aRow1 + k0);
        a1.h[1] = *(const v8u16*)(aRow1 + k0 + 8);
        #pragma unroll
        for (int ct = 0; ct < 8; ++ct) {
            BF16Frag b;
            b.u16 = *(const v16u16*)(RhsT +
                     (size_t)(colBase + ct * 16 + l15) * N_NODES + k0 + half * 16);
            acc0[ct] = __builtin_amdgcn_wmma_f32_16x16x32_bf16(
                false, a0.bf, false, b.bf, (short)0, acc0[ct], false, false);
            acc1[ct] = __builtin_amdgcn_wmma_f32_16x16x32_bf16(
                false, a1.bf, false, b.bf, (short)0, acc1[ct], false, false);
        }
    }

    #pragma unroll
    for (int ct = 0; ct < 8; ++ct) {
        const int col = colBase + ct * 16 + l15;
        #pragma unroll
        for (int r = 0; r < 8; ++r) {
            const int row0 = rowBase + r + half * 8;
            const float v0 = acc0[ct][r];
            Y[(size_t)row0 * JCOLS + col] = v0;
            YbfT[(size_t)col * N_NODES + row0] = f2bf(v0);
            const int row1 = row0 + 16;
            const float v1 = acc1[ct][r];
            Y[(size_t)row1 * JCOLS + col] = v1;
            YbfT[(size_t)col * N_NODES + row1] = f2bf(v1);
        }
    }
}

// ---------------------------------------------------------------------------
// Kernel 6: per-node finale. Block = node n (128 thr = 4 waves).
//   XG[b][r] = bf16 of {X, Y1, 2*Y2 - X}[b, n, i]  staged in LDS (6KB),
//   B operand read straight from WbfT[n] (contiguous 32B per lane).
//   wave w: one 16x16 output tile (cols w*16..+15), 6 WMMAs over K=192.
//   out[b,n,o] = acc + (E[n] . bias_pool[:,o])
// ---------------------------------------------------------------------------
__global__ __launch_bounds__(128) void final_node(
        const float* __restrict__ X, const float* __restrict__ E,
        const unsigned short* __restrict__ WbfT,
        const float* __restrict__ biasPool,
        const float* __restrict__ Y1, const float* __restrict__ Y2,
        float* __restrict__ out) {
    const int n   = blockIdx.x;
    const int tid = threadIdx.x;
    __shared__ unsigned short ldsA[BATCH * RDIM];   // [16][192] bf16
    __shared__ float sE[EMB_D];
    if (tid < EMB_D) sE[tid] = E[n * EMB_D + tid];

    // X_G into LDS
    for (int idx = tid; idx < BATCH * RDIM; idx += 128) {
        const int b = idx / RDIM;
        const int r = idx - b * RDIM;
        const int k = r >> 6;
        const int i = r & 63;
        const float xv = X[((size_t)b * N_NODES + n) * CH + i];
        float v;
        if (k == 0)      v = xv;
        else if (k == 1) v = Y1[(size_t)n * JCOLS + b * CH + i];
        else             v = 2.0f * Y2[(size_t)n * JCOLS + b * CH + i] - xv;
        ldsA[idx] = f2bf(v);
    }
    __syncthreads();

    const int lane = tid & 31;
    const int w    = tid >> 5;
    const int half = lane >> 4;
    const int l15  = lane & 15;
    const int o    = w * 16 + l15;
    const unsigned short* wRow = WbfT + (size_t)n * WCOLS + o * RDIM + half * 16;

    v8f acc = {};
    #pragma unroll
    for (int kk = 0; kk < 6; ++kk) {
        const int k0 = kk * 32;
        BF16Frag a, b;
        const unsigned short* ap = ldsA + l15 * RDIM + k0 + half * 16;
        a.h[0] = *(const v8u16*)(ap);
        a.h[1] = *(const v8u16*)(ap + 8);
        b.u16  = *(const v16u16*)(wRow + k0);
        acc = __builtin_amdgcn_wmma_f32_16x16x32_bf16(
            false, a.bf, false, b.bf, (short)0, acc, false, false);
    }

    float bias = 0.f;
    #pragma unroll
    for (int d = 0; d < EMB_D; ++d) bias += sE[d] * biasPool[d * CH + o];

    #pragma unroll
    for (int r = 0; r < 8; ++r) {
        const int b = r + half * 8;
        out[((size_t)b * N_NODES + n) * CH + o] = acc[r] + bias;
    }
}

// ---------------------------------------------------------------------------
extern "C" void kernel_launch(void* const* d_in, const int* in_sizes, int n_in,
                              void* d_out, int out_size, void* d_ws, size_t ws_size,
                              hipStream_t stream) {
    const float* X        = (const float*)d_in[0];  // [16,4096,64]
    const float* E        = (const float*)d_in[1];  // [4096,16]
    const float* pool     = (const float*)d_in[2];  // [16,3,64,64]
    const float* biasPool = (const float*)d_in[3];  // [16,64]
    float* out = (float*)d_out;                     // [16,4096,64]

    char* ws = (char*)d_ws;
    const size_t MB = 1048576;
    unsigned short* Abf   = (unsigned short*)(ws);             //  32 MB: A bf16 [4096][4096]
    unsigned short* XrT   = (unsigned short*)(ws + 32 * MB);   //   8 MB: Xr^T bf16 [1024][4096]
    float*          Y1    = (float*)(ws + 40 * MB);            //  16 MB: Y1 f32 [4096][1024]
    unsigned short* Y1bfT = (unsigned short*)(ws + 56 * MB);   //   8 MB: Y1^T bf16
    float*          Y2    = (float*)(ws + 64 * MB);            //  16 MB: Y2 f32
    unsigned short* WbfT  = (unsigned short*)(ws + 80 * MB);   //  96 MB: W^T bf16 [4096][12288]
    unsigned short* poolT = (unsigned short*)(ws + 176 * MB);  // 384 KB: pool^T bf16 [12288][16]
    unsigned short* Y2bfT = XrT;                               // dead after pass 1: reuse

    transpose_pool<<<(WCOLS * EMB_D) / 256, 256, 0, stream>>>(pool, poolT);
    wgemm_bf16<<<dim3(WCOLS / 1024, N_NODES / 16), 256, 0, stream>>>(E, poolT, WbfT);
    row_softmax_bf16<<<N_NODES, 256, 0, stream>>>(E, Abf);
    permute_x<<<(N_NODES * JCOLS) / 256, 256, 0, stream>>>(X, XrT);
    gemm_bf16<<<dim3(JCOLS / 128, N_NODES / 128), 128, 0, stream>>>(Abf, XrT, Y1, Y1bfT);
    gemm_bf16<<<dim3(JCOLS / 128, N_NODES / 128), 128, 0, stream>>>(Abf, Y1bfT, Y2, Y2bfT);
    final_node<<<N_NODES, 128, 0, stream>>>(X, E, WbfT, biasPool, Y1, Y2, out);
}
